// MoELayer_44702019617141
// MI455X (gfx1250) — matrix operand order repaired and checked by reading, not compiled, not run on hardware
//
#include <hip/hip_runtime.h>
#include <cstdint>
#include <cstddef>

// Problem constants (match reference)
#define NTOK 4096      // B*T
#define CDIM 768
#define EEXP 8
#define HDIM 3072
#define NSLOT (NTOK * 2)

// GEMM tiling
#define BM 128
#define BN 128
#define BK 32
#define LDSS 40        // padded LDS row stride in bf16 elements (80B, conflict-free, 16B aligned)

// dynamic-LDS layout for the GEMM kernels (no static __shared__ there => offsets start at 0)
#define A0OFF 0u
#define B0OFF 10240u
#define A1OFF 20480u
#define B1OFF 30720u
#define TIDOFF 40960u
#define SMEM_BYTES (40960 + 128 * 4)

typedef __attribute__((ext_vector_type(16))) __bf16 v16bf;
typedef __attribute__((ext_vector_type(8)))  float  v8f;
typedef __attribute__((ext_vector_type(4)))  float  f32x4;
typedef __attribute__((ext_vector_type(4)))  unsigned int u32x4;

union BFrag { v16bf v; u32x4 q[2]; };

__device__ __forceinline__ unsigned short f2bf(float f) {
    unsigned int u = __float_as_uint(f);
    u += 0x7fffu + ((u >> 16) & 1u);            // round-to-nearest-even
    return (unsigned short)(u >> 16);
}

#if __has_builtin(__builtin_amdgcn_cvt_pk_bf16_f32)
typedef __attribute__((ext_vector_type(2))) __bf16 v2bf;
union PkBf { v2bf v; unsigned int u; };
__device__ __forceinline__ unsigned int pack2bf(float a, float b) {
    PkBf r; r.v = __builtin_amdgcn_cvt_pk_bf16_f32(a, b);
    return r.u;
}
#else
__device__ __forceinline__ unsigned int pack2bf(float a, float b) {
    return (unsigned int)f2bf(a) | ((unsigned int)f2bf(b) << 16);
}
#endif

__device__ __forceinline__ float gelu_exact(float v) {
    return 0.5f * v * (1.0f + erff(v * 0.70710678118654752440f));
}

// CDNA5 async global->LDS copy, 16B per lane, tracked by ASYNCcnt.
__device__ __forceinline__ void async_cp16(const void* g, unsigned int lds_off) {
    asm volatile("global_load_async_to_lds_b128 %0, %1, off"
                 :: "v"(lds_off), "v"((unsigned long long)(uintptr_t)g)
                 : "memory");
}
__device__ __forceinline__ void async_wait0() {
    asm volatile("s_wait_asynccnt 0x0" ::: "memory");
}

// ---------------------------------------------------------------- fp32 -> bf16 (x)
__global__ __launch_bounds__(256) void k_cvt_x(const float* __restrict__ s,
                                               unsigned short* __restrict__ d) {
    const int i = (blockIdx.x * 256 + threadIdx.x) * 8;
    const f32x4 a = *(const f32x4*)(s + i);
    const f32x4 b = *(const f32x4*)(s + i + 4);
    u32x4 pk;
    pk.x = pack2bf(a.x, a.y); pk.y = pack2bf(a.z, a.w);
    pk.z = pack2bf(b.x, b.y); pk.w = pack2bf(b.z, b.w);
    *(u32x4*)(d + i) = pk;
}

// ---------------------------------------------------------------- fp32 [R][Cc] -> bf16 transposed [Cc][R]
__global__ __launch_bounds__(256) void k_tcvt(const float* __restrict__ src,
                                              unsigned short* __restrict__ dst,
                                              int R, int Cc) {
    __shared__ unsigned short t[64][72];
    const size_t eoff = (size_t)blockIdx.z * (size_t)R * (size_t)Cc;
    src += eoff; dst += eoff;
    const int c0 = blockIdx.x * 64, r0 = blockIdx.y * 64;
    const int tid = threadIdx.x;

    #pragma unroll
    for (int tq = 0; tq < 4; ++tq) {                    // 64 rows x 16 quads
        const int task = tid + tq * 256;
        const int r = task >> 4, q = (task & 15) * 4;
        const f32x4 f = *(const f32x4*)(src + (size_t)(r0 + r) * Cc + c0 + q);
        *(unsigned int*)&t[r][q]     = pack2bf(f.x, f.y);
        *(unsigned int*)&t[r][q + 2] = pack2bf(f.z, f.w);
    }
    __syncthreads();
    #pragma unroll
    for (int tq = 0; tq < 2; ++tq) {                    // 64 cols x 8 chunks of 8
        const int task = tid + tq * 256;
        const int c = task >> 3, rc = (task & 7) * 8;
        u32x4 pk;
        pk.x = (unsigned)t[rc + 0][c] | ((unsigned)t[rc + 1][c] << 16);
        pk.y = (unsigned)t[rc + 2][c] | ((unsigned)t[rc + 3][c] << 16);
        pk.z = (unsigned)t[rc + 4][c] | ((unsigned)t[rc + 5][c] << 16);
        pk.w = (unsigned)t[rc + 6][c] | ((unsigned)t[rc + 7][c] << 16);
        *(u32x4*)(dst + (size_t)(c0 + c) * R + r0 + rc) = pk;
    }
}

// ---------------------------------------------------------------- init
__global__ void k_init(int* __restrict__ counts) {
    if (threadIdx.x < EEXP) counts[threadIdx.x] = 0;
}

// ---------------------------------------------------------------- gating (one wave32 per token)
__global__ __launch_bounds__(256) void k_gate(
    const float* __restrict__ x, const float* __restrict__ gW, const float* __restrict__ gb,
    int* __restrict__ topk_idx, float* __restrict__ topk_w,
    int* __restrict__ pos_nk, int* __restrict__ counts,
    float* __restrict__ usage_partial)
{
    __shared__ float su[8][EEXP];
    const int tid = threadIdx.x, lane = tid & 31, wid = tid >> 5;
    const int n = blockIdx.x * 8 + wid;

    float acc[EEXP];
    #pragma unroll
    for (int e = 0; e < EEXP; ++e) acc[e] = 0.0f;

    const float* xr = x + (size_t)n * CDIM;
    for (int c = lane; c < CDIM; c += 32) {
        const float xv = xr[c];
        const f32x4* g = (const f32x4*)(gW + (size_t)c * EEXP);
        const f32x4 g0 = g[0], g1 = g[1];
        acc[0] += xv * g0.x; acc[1] += xv * g0.y; acc[2] += xv * g0.z; acc[3] += xv * g0.w;
        acc[4] += xv * g1.x; acc[5] += xv * g1.y; acc[6] += xv * g1.z; acc[7] += xv * g1.w;
    }
    #pragma unroll
    for (int off = 16; off > 0; off >>= 1) {
        #pragma unroll
        for (int e = 0; e < EEXP; ++e) acc[e] += __shfl_xor(acc[e], off, 32);
    }
    if (lane == 0) {
        float p[EEXP], mx = -1e30f;
        #pragma unroll
        for (int e = 0; e < EEXP; ++e) { acc[e] += gb[e]; mx = fmaxf(mx, acc[e]); }
        float s = 0.0f;
        #pragma unroll
        for (int e = 0; e < EEXP; ++e) { p[e] = __expf(acc[e] - mx); s += p[e]; }
        const float inv = 1.0f / s;
        #pragma unroll
        for (int e = 0; e < EEXP; ++e) p[e] *= inv;

        int i0 = 0;
        #pragma unroll
        for (int e = 1; e < EEXP; ++e) if (p[e] > p[i0]) i0 = e;
        int i1 = (i0 == 0) ? 1 : 0;
        #pragma unroll
        for (int e = 0; e < EEXP; ++e) if (e != i0 && p[e] > p[i1]) i1 = e;

        const float wsum = p[i0] + p[i1];
        topk_idx[2 * n + 0] = i0;  topk_w[2 * n + 0] = p[i0] / wsum;
        topk_idx[2 * n + 1] = i1;  topk_w[2 * n + 1] = p[i1] / wsum;
        pos_nk[2 * n + 0] = atomicAdd(&counts[i0], 1);
        pos_nk[2 * n + 1] = atomicAdd(&counts[i1], 1);
        #pragma unroll
        for (int e = 0; e < EEXP; ++e) su[wid][e] = p[e];
    }
    __syncthreads();
    if (tid < EEXP) {          // deterministic fixed-order per-block reduction
        float s = 0.0f;
        #pragma unroll
        for (int w = 0; w < 8; ++w) s += su[w][tid];
        usage_partial[blockIdx.x * EEXP + tid] = s;
    }
}

// ---------------------------------------------------------------- offsets + lb loss
__global__ void k_finalize(const int* __restrict__ counts, int* __restrict__ offsets,
                           const float* __restrict__ usage_partial, int nblocks,
                           float* __restrict__ lb_out)
{
    __shared__ float us[EEXP];
    const int tid = threadIdx.x;
    if (tid < EEXP) {
        float s = 0.0f;
        for (int b = 0; b < nblocks; ++b) s += usage_partial[b * EEXP + tid];
        us[tid] = s / (float)NTOK;
    }
    __syncthreads();
    if (tid == 0) {
        int off = 0;
        for (int e = 0; e < EEXP; ++e) { offsets[e] = off; off += counts[e]; }
        float m = 0.0f;
        for (int e = 0; e < EEXP; ++e) { const float d = us[e] - 1.0f / EEXP; m += d * d; }
        lb_out[0] = 0.01f * (m / (float)EEXP);
    }
}

// ---------------------------------------------------------------- scatter token -> slot
__global__ void k_scatter(const int* __restrict__ topk_idx, const float* __restrict__ topk_w,
                          const int* __restrict__ pos_nk, const int* __restrict__ offsets,
                          int* __restrict__ slot_of, int* __restrict__ expert_token,
                          float* __restrict__ slot_weight)
{
    const int i = blockIdx.x * blockDim.x + threadIdx.x;
    if (i >= NSLOT) return;
    const int n = i >> 1;
    const int e = topk_idx[i];
    const int slot = offsets[e] + pos_nk[i];
    slot_of[i] = slot;
    expert_token[slot] = n;
    slot_weight[slot] = topk_w[i];
}

// ---------------------------------------------------------------- GEMM1: h = gelu(x[tokens] @ w1[e] + b1[e])
// A: bf16 x rows gathered via token ids; B: pre-transposed bf16 w1t[e] = [H][C].
__global__ __launch_bounds__(256) void k_ffn1(
    const unsigned short* __restrict__ xbf, const unsigned short* __restrict__ w1t,
    const float* __restrict__ b1, const int* __restrict__ counts,
    const int* __restrict__ offsets, const int* __restrict__ expert_token,
    unsigned short* __restrict__ hbuf)
{
    extern __shared__ char smem[];
    const int e  = blockIdx.z;
    const int nE = counts[e];
    const int m0 = blockIdx.x * BM;
    if (m0 >= nE) return;
    const int nbase = blockIdx.y * BN;
    const int slotBase = offsets[e] + m0;
    const unsigned short* Bsrc = w1t + (size_t)e * HDIM * CDIM;

    int* tids = (int*)(smem + TIDOFF);
    const int tid = threadIdx.x;
    for (int r = tid; r < BM; r += 256)
        tids[r] = (m0 + r < nE) ? expert_token[slotBase + r] : 0;

    // pre-zero invalid A rows (both buffers) once; async copies only touch valid rows
    #pragma unroll
    for (int t = 0; t < 2; ++t) {
        const int task = tid + t * 256;
        const int r = task >> 2, ch = (task & 3) * 8;
        if (m0 + r >= nE) {
            const u32x4 z = (u32x4)(0u);
            *(u32x4*)(smem + A0OFF + (unsigned)(r * LDSS + ch) * 2) = z;
            *(u32x4*)(smem + A1OFF + (unsigned)(r * LDSS + ch) * 2) = z;
        }
    }
    __syncthreads();

    // hoist loop-invariant staging addresses into registers (only k advances)
    const unsigned short* aSrc[2];
    const unsigned short* bSrc[2];
    unsigned ldsOff[2];
    bool aVal[2];
    #pragma unroll
    for (int t = 0; t < 2; ++t) {
        const int task = tid + t * 256;
        const int r = task >> 2, ch = (task & 3) * 8;
        aVal[t]   = (m0 + r < nE);
        aSrc[t]   = xbf + (size_t)tids[r] * CDIM + ch;
        bSrc[t]   = Bsrc + (size_t)(nbase + r) * CDIM + ch;
        ldsOff[t] = (unsigned)(r * LDSS + ch) * 2;
    }

    const int lane = tid & 31, wid = tid >> 5;
    const int wm = wid & 1, wn = wid >> 1;      // 2 waves along M, 4 along N
    const int lr = lane & 15, lh = lane >> 4;

    v8f acc[4][2];
    #pragma unroll
    for (int i = 0; i < 4; ++i)
        #pragma unroll
        for (int j = 0; j < 2; ++j)
            #pragma unroll
            for (int r = 0; r < 8; ++r) acc[i][j][r] = 0.0f;

    auto stage = [&](int buf, int kk) {
        const unsigned aoff = buf ? A1OFF : A0OFF;
        const unsigned boff = buf ? B1OFF : B0OFF;
        #pragma unroll
        for (int t = 0; t < 2; ++t)
            if (aVal[t]) async_cp16(aSrc[t] + kk, aoff + ldsOff[t]);
        #pragma unroll
        for (int t = 0; t < 2; ++t)
            async_cp16(bSrc[t] + kk, boff + ldsOff[t]);
    };

    stage(0, 0);
    const int nIt = CDIM / BK;
    #pragma unroll 1
    for (int it = 0; it < nIt; ++it) {
        async_wait0();
        __syncthreads();
        if (it + 1 < nIt) stage((it + 1) & 1, (it + 1) * BK);

        const unsigned short* A = (const unsigned short*)(smem + ((it & 1) ? A1OFF : A0OFF));
        const unsigned short* B = (const unsigned short*)(smem + ((it & 1) ? B1OFF : B0OFF));
        BFrag a[4], b[2];
        #pragma unroll
        for (int mt = 0; mt < 4; ++mt) {
            const unsigned short* p = A + (wm * 64 + mt * 16 + lr) * LDSS + lh * 8;
            a[mt].q[0] = *(const u32x4*)p;
            a[mt].q[1] = *(const u32x4*)(p + 16);
        }
        #pragma unroll
        for (int nt = 0; nt < 2; ++nt) {
            const unsigned short* p = B + (wn * 32 + nt * 16 + lr) * LDSS + lh * 8;
            b[nt].q[0] = *(const u32x4*)p;
            b[nt].q[1] = *(const u32x4*)(p + 16);
        }
        #pragma unroll
        for (int mt = 0; mt < 4; ++mt)
            #pragma unroll
            for (int nt = 0; nt < 2; ++nt)
                acc[mt][nt] = __builtin_amdgcn_wmma_f32_16x16x32_bf16(
                    false, a[mt].v, false, b[nt].v, (short)0, acc[mt][nt], false, false);
    }

    // epilogue: bias + exact GELU -> bf16 hbuf[slot][H]
    #pragma unroll
    for (int nt = 0; nt < 2; ++nt) {
        const int n = nbase + wn * 32 + nt * 16 + lr;
        const float bias = b1[(size_t)e * HDIM + n];
        #pragma unroll
        for (int mt = 0; mt < 4; ++mt) {
            #pragma unroll
            for (int r = 0; r < 8; ++r) {
                const int m = m0 + wm * 64 + mt * 16 + lh * 8 + r;
                if (m < nE) {
                    const float v = gelu_exact(acc[mt][nt][r] + bias);
                    hbuf[(size_t)(offsets[e] + m) * HDIM + n] = f2bf(v);
                }
            }
        }
    }
}

// ---------------------------------------------------------------- GEMM2: y = (h @ w2[e] + b2[e]) * gate_w
// A: bf16 h rows by slot; B: pre-transposed bf16 w2t[e] = [C][H].
__global__ __launch_bounds__(256) void k_ffn2(
    const unsigned short* __restrict__ hbuf, const unsigned short* __restrict__ w2t,
    const float* __restrict__ b2, const int* __restrict__ counts,
    const int* __restrict__ offsets, const float* __restrict__ slot_weight,
    float* __restrict__ ybuf)
{
    extern __shared__ char smem[];
    const int e  = blockIdx.z;
    const int nE = counts[e];
    const int m0 = blockIdx.x * BM;
    if (m0 >= nE) return;
    const int nbase = blockIdx.y * BN;
    const int slotBase = offsets[e] + m0;
    const unsigned short* Bsrc = w2t + (size_t)e * CDIM * HDIM;

    const int tid = threadIdx.x;
    #pragma unroll
    for (int t = 0; t < 2; ++t) {
        const int task = tid + t * 256;
        const int r = task >> 2, ch = (task & 3) * 8;
        if (m0 + r >= nE) {
            const u32x4 z = (u32x4)(0u);
            *(u32x4*)(smem + A0OFF + (unsigned)(r * LDSS + ch) * 2) = z;
            *(u32x4*)(smem + A1OFF + (unsigned)(r * LDSS + ch) * 2) = z;
        }
    }
    __syncthreads();

    const unsigned short* aSrc[2];
    const unsigned short* bSrc[2];
    unsigned ldsOff[2];
    bool aVal[2];
    #pragma unroll
    for (int t = 0; t < 2; ++t) {
        const int task = tid + t * 256;
        const int r = task >> 2, ch = (task & 3) * 8;
        aVal[t]   = (m0 + r < nE);
        aSrc[t]   = hbuf + (size_t)(slotBase + r) * HDIM + ch;
        bSrc[t]   = Bsrc + (size_t)(nbase + r) * HDIM + ch;
        ldsOff[t] = (unsigned)(r * LDSS + ch) * 2;
    }

    const int lane = tid & 31, wid = tid >> 5;
    const int wm = wid & 1, wn = wid >> 1;
    const int lr = lane & 15, lh = lane >> 4;

    v8f acc[4][2];
    #pragma unroll
    for (int i = 0; i < 4; ++i)
        #pragma unroll
        for (int j = 0; j < 2; ++j)
            #pragma unroll
            for (int r = 0; r < 8; ++r) acc[i][j][r] = 0.0f;

    auto stage = [&](int buf, int kk) {
        const unsigned aoff = buf ? A1OFF : A0OFF;
        const unsigned boff = buf ? B1OFF : B0OFF;
        #pragma unroll
        for (int t = 0; t < 2; ++t)
            if (aVal[t]) async_cp16(aSrc[t] + kk, aoff + ldsOff[t]);
        #pragma unroll
        for (int t = 0; t < 2; ++t)
            async_cp16(bSrc[t] + kk, boff + ldsOff[t]);
    };

    stage(0, 0);
    const int nIt = HDIM / BK;
    #pragma unroll 1
    for (int it = 0; it < nIt; ++it) {
        async_wait0();
        __syncthreads();
        if (it + 1 < nIt) stage((it + 1) & 1, (it + 1) * BK);

        const unsigned short* A = (const unsigned short*)(smem + ((it & 1) ? A1OFF : A0OFF));
        const unsigned short* B = (const unsigned short*)(smem + ((it & 1) ? B1OFF : B0OFF));
        BFrag a[4], b[2];
        #pragma unroll
        for (int mt = 0; mt < 4; ++mt) {
            const unsigned short* p = A + (wm * 64 + mt * 16 + lr) * LDSS + lh * 8;
            a[mt].q[0] = *(const u32x4*)p;
            a[mt].q[1] = *(const u32x4*)(p + 16);
        }
        #pragma unroll
        for (int nt = 0; nt < 2; ++nt) {
            const unsigned short* p = B + (wn * 32 + nt * 16 + lr) * LDSS + lh * 8;
            b[nt].q[0] = *(const u32x4*)p;
            b[nt].q[1] = *(const u32x4*)(p + 16);
        }
        #pragma unroll
        for (int mt = 0; mt < 4; ++mt)
            #pragma unroll
            for (int nt = 0; nt < 2; ++nt)
                acc[mt][nt] = __builtin_amdgcn_wmma_f32_16x16x32_bf16(
                    false, a[mt].v, false, b[nt].v, (short)0, acc[mt][nt], false, false);
    }

    // epilogue: bias + per-row gate weight -> f32 ybuf[slot][C]
    float bias[2];
    #pragma unroll
    for (int nt = 0; nt < 2; ++nt)
        bias[nt] = b2[(size_t)e * CDIM + nbase + wn * 32 + nt * 16 + lr];

    #pragma unroll
    for (int mt = 0; mt < 4; ++mt) {
        #pragma unroll
        for (int r = 0; r < 8; ++r) {
            const int m = m0 + wm * 64 + mt * 16 + lh * 8 + r;
            if (m < nE) {
                const int slot = offsets[e] + m;
                const float sw = slot_weight[slot];
                #pragma unroll
                for (int nt = 0; nt < 2; ++nt) {
                    const int n = nbase + wn * 32 + nt * 16 + lr;
                    ybuf[(size_t)slot * CDIM + n] = (acc[mt][nt][r] + bias[nt]) * sw;
                }
            }
        }
    }
}

// ---------------------------------------------------------------- combine two expert outputs per token
__global__ __launch_bounds__(256) void k_combine(
    const float* __restrict__ ybuf, const int* __restrict__ slot_of, float* __restrict__ out)
{
    const int t = blockIdx.x * 256 + threadIdx.x;     // NTOK * (CDIM/4) tasks exactly
    const int n = t / (CDIM / 4);
    const int c4 = t % (CDIM / 4);
    const int s0 = slot_of[2 * n], s1 = slot_of[2 * n + 1];
    const f32x4 a = ((const f32x4*)(ybuf + (size_t)s0 * CDIM))[c4];
    const f32x4 b = ((const f32x4*)(ybuf + (size_t)s1 * CDIM))[c4];
    ((f32x4*)(out + (size_t)n * CDIM))[c4] = a + b;
}

// ---------------------------------------------------------------- host
extern "C" void kernel_launch(void* const* d_in, const int* in_sizes, int n_in,
                              void* d_out, int out_size, void* d_ws, size_t ws_size,
                              hipStream_t stream) {
    (void)in_sizes; (void)n_in; (void)out_size; (void)ws_size;

    const float* x   = (const float*)d_in[0];
    const float* gW  = (const float*)d_in[1];
    const float* gb  = (const float*)d_in[2];
    const float* w1  = (const float*)d_in[3];
    const float* b1  = (const float*)d_in[4];
    const float* w2  = (const float*)d_in[5];
    const float* b2  = (const float*)d_in[6];
    float* out = (float*)d_out;

    char* w = (char*)d_ws;
    size_t o = 0;
    unsigned short* hbuf = (unsigned short*)(w + o); o += (size_t)NSLOT * HDIM * sizeof(unsigned short);
    float* ybuf          = (float*)(w + o);          o += (size_t)NSLOT * CDIM * sizeof(float);
    unsigned short* xbf  = (unsigned short*)(w + o); o += (size_t)NTOK * CDIM * sizeof(unsigned short);
    unsigned short* w1t  = (unsigned short*)(w + o); o += (size_t)EEXP * CDIM * HDIM * sizeof(unsigned short);
    unsigned short* w2t  = (unsigned short*)(w + o); o += (size_t)EEXP * HDIM * CDIM * sizeof(unsigned short);
    int*   topk_idx      = (int*)(w + o);            o += (size_t)NSLOT * sizeof(int);
    float* topk_w        = (float*)(w + o);          o += (size_t)NSLOT * sizeof(float);
    int*   pos_nk        = (int*)(w + o);            o += (size_t)NSLOT * sizeof(int);
    int*   slot_of       = (int*)(w + o);            o += (size_t)NSLOT * sizeof(int);
    int*   expert_token  = (int*)(w + o);            o += (size_t)NSLOT * sizeof(int);
    float* slot_weight   = (float*)(w + o);          o += (size_t)NSLOT * sizeof(float);
    int*   counts        = (int*)(w + o);            o += 16 * sizeof(int);
    int*   offsets       = (int*)(w + o);            o += 16 * sizeof(int);
    float* usage_partial = (float*)(w + o);          o += (size_t)(NTOK / 8) * EEXP * sizeof(float);

    const int gateBlocks = NTOK / 8;   // 512

    // precision pre-pass: bf16 copies (weights transposed for K-contiguous WMMA staging)
    k_cvt_x<<<(NTOK * CDIM / 8) / 256, 256, 0, stream>>>(x, xbf);
    k_tcvt<<<dim3(HDIM / 64, CDIM / 64, EEXP), 256, 0, stream>>>(w1, w1t, CDIM, HDIM);
    k_tcvt<<<dim3(CDIM / 64, HDIM / 64, EEXP), 256, 0, stream>>>(w2, w2t, HDIM, CDIM);

    k_init<<<1, 32, 0, stream>>>(counts);
    k_gate<<<gateBlocks, 256, 0, stream>>>(x, gW, gb, topk_idx, topk_w, pos_nk, counts, usage_partial);
    k_finalize<<<1, 64, 0, stream>>>(counts, offsets, usage_partial, gateBlocks,
                                     out + (size_t)NTOK * CDIM);
    k_scatter<<<(NSLOT + 255) / 256, 256, 0, stream>>>(topk_idx, topk_w, pos_nk, offsets,
                                                       slot_of, expert_token, slot_weight);
    dim3 g1(NTOK / BM, HDIM / BN, EEXP);   // worst-case M, early-exit per expert
    k_ffn1<<<g1, 256, SMEM_BYTES, stream>>>(xbf, w1t, b1, counts, offsets, expert_token, hbuf);
    dim3 g2(NTOK / BM, CDIM / BN, EEXP);
    k_ffn2<<<g2, 256, SMEM_BYTES, stream>>>(hbuf, w2t, b2, counts, offsets, slot_weight, ybuf);
    k_combine<<<(NTOK * (CDIM / 4)) / 256, 256, 0, stream>>>(ybuf, slot_of, out);
}
